// LkyMultiheadAttention_68676527063522
// MI455X (gfx1250) — matrix operand order repaired and checked by reading, not compiled, not run on hardware
//
#include <hip/hip_runtime.h>
#include <hip/hip_bf16.h>

#define SEQ    2048
#define DMODEL 1024
#define NHEAD  16
#define HD     64

typedef __attribute__((ext_vector_type(16))) __bf16 v16bf;
typedef __attribute__((ext_vector_type(8)))  float  v8f;

union Frag16 { unsigned u[8]; uint4 q[2]; v16bf v; };

__device__ __forceinline__ unsigned f2bf(float f) {
  unsigned u = __builtin_bit_cast(unsigned, f);
  return (u + 0x7fffu + ((u >> 16) & 1u)) >> 16;   // round-to-nearest-even
}

__device__ __forceinline__ v8f wmma_bf16(const Frag16& a, const Frag16& b, v8f c) {
  return __builtin_amdgcn_wmma_f32_16x16x32_bf16(false, a.v, false, b.v,
                                                 (short)0, c, false, false);
}

// ---- optional gfx1250 async global->LDS path (ASYNCcnt) -------------------
#if defined(__has_builtin)
#if __has_builtin(__builtin_amdgcn_global_load_async_to_lds_b32) && \
    __has_builtin(__builtin_amdgcn_global_load_async_to_lds_b128)
#define HAVE_ASYNC_LDS 1
#endif
#if __has_builtin(__builtin_amdgcn_s_wait_asynccnt)
#define HAVE_WAIT_ASYNC 1
#endif
#endif

#if HAVE_ASYNC_LDS
typedef int v4i_vs __attribute__((vector_size(16)));
typedef __attribute__((address_space(1))) int*     ga_i32p;   // global src (b32)
typedef __attribute__((address_space(1))) v4i_vs*  ga_v4ip;   // global src (b128)
typedef __attribute__((address_space(3))) int*     la_i32p;   // LDS dst (b32)
typedef __attribute__((address_space(3))) v4i_vs*  la_v4ip;   // LDS dst (b128)
__device__ __forceinline__ void acp_b32(unsigned* l, const unsigned* g) {
  __builtin_amdgcn_global_load_async_to_lds_b32((ga_i32p)g, (la_i32p)l, 0, 0);
}
__device__ __forceinline__ void acp_b128(unsigned* l, const unsigned* g) {
  __builtin_amdgcn_global_load_async_to_lds_b128((ga_v4ip)g, (la_v4ip)l, 0, 0);
}
__device__ __forceinline__ void async_wait0() {
#if HAVE_WAIT_ASYNC
  __builtin_amdgcn_s_wait_asynccnt(0);
#else
  asm volatile("s_wait_asynccnt 0" ::: "memory");
#endif
}
#endif

// ---------------------------------------------------------------------------
// One-time packers: f32 -> bf16.
// ---------------------------------------------------------------------------
__global__ __launch_bounds__(256) void pack_rows_kernel(
    const float* __restrict__ in, unsigned* __restrict__ out, int npairs) {
  int i = blockIdx.x * 256 + threadIdx.x;
  if (i >= npairs) return;
  float2 v = ((const float2*)in)[i];
  out[i] = f2bf(v.x) | (f2bf(v.y) << 16);
}

__global__ __launch_bounds__(256) void pack_weight_kernel(
    const float* __restrict__ w, unsigned* __restrict__ wp, int total) {
  int i = blockIdx.x * 256 + threadIdx.x;
  if (i >= total) return;
  int n = i & (DMODEL - 1);
  int k2 = i >> 10;                       // log2(DMODEL)
  wp[i] = f2bf(w[(size_t)(2 * k2) * DMODEL + n]) |
          (f2bf(w[(size_t)(2 * k2 + 1) * DMODEL + n]) << 16);
}

// ---------------------------------------------------------------------------
// C(MxN) = A(MxK,bf16 row-major) * B(K/2 x N, u32 pair-packed bf16).
// 64x64 tile / 128 threads (4 waves). A fragments: 2x b128 global loads.
// B tile staged in LDS *transposed* (BS[n][k2], stride 20) so each lane's
// B fragment is two contiguous, 16B-aligned b128 LDS loads.
// cmode: 0 = store u32 pair-packed along M (V); 1 = RoPE + packed bf16;
// 2 = store f32.
// ---------------------------------------------------------------------------
__global__ __launch_bounds__(128) void gemm_wmma(
    const unsigned* __restrict__ Au32,
    const unsigned* __restrict__ Bpk,
    void* __restrict__ Cp, const int cmode,
    const int M, const int N, const int K)
{
  __shared__ __align__(16) unsigned smem[4160];            // 16640 B
  unsigned* BS = smem;                                     // 64 x 20 transposed
  float*    CS = (float*)smem;                             // 64 x 65 (epilogue)

  const int tid  = threadIdx.x;
  const int lane = tid & 31, wv = tid >> 5;
  const int ml   = lane & 15, hi = lane >> 4;
  const int m0 = blockIdx.y * 64, n0 = blockIdx.x * 64;

  v8f acc[4];
#pragma unroll
  for (int t = 0; t < 4; ++t)
#pragma unroll
    for (int r = 0; r < 8; ++r) acc[t][r] = 0.0f;

  const size_t arow = (size_t)(m0 + wv * 16 + ml) * (K >> 1);

  for (int k0 = 0; k0 < K; k0 += 32) {
    // stage B tile transposed: BS[n][k2] = Bpk[k0/2+k2][n0+n]
#pragma unroll
    for (int it = 0; it < 8; ++it) {
      int idx = tid + it * 128;
      int n = idx & 63, k2 = idx >> 6;
      const unsigned* g = Bpk + (size_t)((k0 >> 1) + k2) * N + n0 + n;
      unsigned* l = BS + n * 20 + k2;
#if HAVE_ASYNC_LDS
      acp_b32(l, g);
#else
      *l = *g;
#endif
    }
#if HAVE_ASYNC_LDS
    async_wait0();
#endif
    __syncthreads();

    // A fragment: two 16B global loads (pair indices are consecutive)
    Frag16 afr;
    {
      const uint4* ap = (const uint4*)(Au32 + arow + (k0 >> 1) + hi * 4);
      afr.q[0] = ap[0];
      afr.q[1] = ap[2];
    }
#pragma unroll
    for (int t = 0; t < 4; ++t) {
      Frag16 bfr;
      const unsigned* bp = BS + (t * 16 + ml) * 20 + hi * 4;
      bfr.q[0] = *(const uint4*)bp;
      bfr.q[1] = *(const uint4*)(bp + 8);
      acc[t] = wmma_bf16(afr, bfr, acc[t]);
    }
    __syncthreads();
  }

  if (cmode == 1) {
    // RoPE epilogue: pairs are adjacent columns -> round-trip through LDS
#pragma unroll
    for (int t = 0; t < 4; ++t)
#pragma unroll
      for (int r = 0; r < 8; ++r)
        CS[(wv * 16 + r + 8 * hi) * 65 + t * 16 + ml] = acc[t][r];
    __syncthreads();
    unsigned short* ob = (unsigned short*)Cp;
#pragma unroll
    for (int it = 0; it < 16; ++it) {
      int i = tid + it * 128;
      int pr = i & 31, row = i >> 5;
      int gm = m0 + row;
      int s  = gm & (SEQ - 1);
      int c  = n0 + 2 * pr;
      int di = (c & (HD - 1)) >> 1;
      float freq = __expf(-0.28782313662425572f * (float)di); // theta^(-2i/hd)
      float ang  = (float)s * freq;
      float sn, cs;
      __sincosf(ang, &sn, &cs);
      float c0 = CS[row * 65 + 2 * pr];
      float c1 = CS[row * 65 + 2 * pr + 1];
      *(unsigned*)(ob + (size_t)gm * N + c) =
          f2bf(c0 * cs - c1 * sn) | (f2bf(c0 * sn + c1 * cs) << 16);
    }
  } else if (cmode == 0) {
    // pair-pack along M: row pairs (2j,2j+1) live in adjacent VGPRs
    unsigned* vp = (unsigned*)Cp;
#pragma unroll
    for (int t = 0; t < 4; ++t)
#pragma unroll
      for (int j = 0; j < 4; ++j) {
        int rp = ((m0 + wv * 16) >> 1) + hi * 4 + j;
        vp[(size_t)rp * N + n0 + t * 16 + ml] =
            f2bf(acc[t][2 * j]) | (f2bf(acc[t][2 * j + 1]) << 16);
      }
  } else {
    float* of = (float*)Cp;
#pragma unroll
    for (int t = 0; t < 4; ++t)
#pragma unroll
      for (int r = 0; r < 8; ++r) {
        int row = m0 + wv * 16 + r + 8 * hi;
        of[(size_t)row * N + n0 + t * 16 + ml] = acc[t][r];
      }
  }
}

// ---------------------------------------------------------------------------
// Causal flash attention: one block per (head, batch, 64-query tile).
// All LDS tiles stored transposed so every WMMA operand is 2x ds_load_b128.
// ---------------------------------------------------------------------------
__global__ __launch_bounds__(128) void attn_wmma(
    const unsigned* __restrict__ qbu,
    const unsigned* __restrict__ kbu,
    const unsigned* __restrict__ vp,
    unsigned short* __restrict__ ob)
{
  __shared__ __align__(16) unsigned smem[6144];
  unsigned* KB = smem;                       // 32 x 36: KB[key][d2]
  unsigned* VB = KB + 32 * 36;               // 64 x 20: VB[d][k2]
  float*    SW = (float*)(VB + 64 * 20);     // 64 x 36: raw scores
  unsigned* PB = (unsigned*)(SW + 64 * 36);  // 64 x 20: P bf16 pairs (A layout)
  float*    AL = (float*)(PB + 64 * 20);     // 64: per-row alpha
  float*    LL = AL + 64;                    // 64: per-row l

  const int tid  = threadIdx.x;
  const int lane = tid & 31, wv = tid >> 5;
  const int ml   = lane & 15, hi = lane >> 4;
  const int q0 = blockIdx.x * 64;
  const int h  = blockIdx.y >> 1, b = blockIdx.y & 1;
  const size_t base  = (size_t)b * SEQ * DMODEL + (size_t)h * HD;   // bf16 elems
  const size_t baseu = base >> 1;                                   // u32 units
  const size_t vbase = (size_t)b * (SEQ / 2) * DMODEL + (size_t)h * HD;

  // Q fragments in VGPRs for the whole block (two 16B loads per k-step)
  Frag16 qf[2];
  {
    const uint4* qp =
        (const uint4*)(qbu + baseu + (size_t)(q0 + wv * 16 + ml) * (DMODEL / 2) +
                       hi * 4);
#pragma unroll
    for (int f = 0; f < 2; ++f) {
      qf[f].q[0] = qp[f * 4];
      qf[f].q[1] = qp[f * 4 + 2];
    }
  }

  float m_i = -__builtin_inff(), l_i = 0.0f;
  v8f oacc[4];
#pragma unroll
  for (int t = 0; t < 4; ++t)
#pragma unroll
    for (int r = 0; r < 8; ++r) oacc[t][r] = 0.0f;

  const int nblk = (q0 + 64) >> 5;
  for (int blk = 0; blk < nblk; ++blk) {
    const int key0 = blk * 32;
    // stage K block: KB[key][d2], contiguous in both global and LDS -> b128
#pragma unroll
    for (int it = 0; it < 2; ++it) {
      int idx = tid + it * 128;
      int d4 = (idx & 7) * 4, key = idx >> 3;
      const unsigned* g = kbu + baseu + (size_t)(key0 + key) * (DMODEL / 2) + d4;
      unsigned* l = KB + key * 36 + d4;
#if HAVE_ASYNC_LDS
      acp_b128(l, g);
#else
      *(uint4*)l = *(const uint4*)g;
#endif
    }
    // stage V block transposed: VB[d][k2] (pair-packed along key)
#pragma unroll
    for (int it = 0; it < 8; ++it) {
      int idx = tid + it * 128;
      int d = idx & 63, k2 = idx >> 6;
      const unsigned* g = vp + vbase + (size_t)((key0 >> 1) + k2) * DMODEL + d;
      unsigned* l = VB + d * 20 + k2;
#if HAVE_ASYNC_LDS
      acp_b32(l, g);
#else
      *l = *g;
#endif
    }
#if HAVE_ASYNC_LDS
    async_wait0();
#endif
    __syncthreads();

    // S = Q (16x64) @ K^T (64x32): 2 key tiles x 2 k-steps = 4 WMMA
#pragma unroll
    for (int t = 0; t < 2; ++t) {
      v8f sacc;
#pragma unroll
      for (int r = 0; r < 8; ++r) sacc[r] = 0.0f;
#pragma unroll
      for (int f = 0; f < 2; ++f) {
        Frag16 kf;
        const unsigned* kp = KB + (t * 16 + ml) * 36 + f * 16 + hi * 4;
        kf.q[0] = *(const uint4*)kp;
        kf.q[1] = *(const uint4*)(kp + 8);
        sacc = wmma_bf16(qf[f], kf, sacc);
      }
#pragma unroll
      for (int r = 0; r < 8; ++r)
        SW[(wv * 16 + r + 8 * hi) * 36 + t * 16 + ml] = sacc[r];
    }

    // online softmax: one query row per lane (lanes 0..15 of each wave)
    if (lane < 16) {
      const int qg = q0 + wv * 16 + lane;
      const float* srow = SW + (wv * 16 + lane) * 36;
      float sv[32];
      float bmax = -__builtin_inff();
#pragma unroll
      for (int j = 0; j < 32; ++j) {
        float s = srow[j] * 0.125f;               // 1/sqrt(64)
        if (key0 + j > qg) s = -__builtin_inff(); // causal mask
        sv[j] = s;
        bmax = fmaxf(bmax, s);
      }
      const float nm    = fmaxf(m_i, bmax);
      const float alpha = __expf(m_i - nm);
      float sum = 0.0f;
      unsigned* prow = PB + (wv * 16 + lane) * 20;
#pragma unroll
      for (int j2 = 0; j2 < 16; ++j2) {
        float p0 = __expf(sv[2 * j2]     - nm);
        float p1 = __expf(sv[2 * j2 + 1] - nm);
        sum += p0 + p1;
        prow[j2] = f2bf(p0) | (f2bf(p1) << 16);
      }
      l_i = l_i * alpha + sum;
      m_i = nm;
      AL[wv * 16 + lane] = alpha;
    }

    // rescale running O, then O += P (16x32) @ V (32x64): 4 WMMA
    float av[8];
#pragma unroll
    for (int r = 0; r < 8; ++r) av[r] = AL[wv * 16 + r + 8 * hi];
#pragma unroll
    for (int t = 0; t < 4; ++t)
#pragma unroll
      for (int r = 0; r < 8; ++r) oacc[t][r] *= av[r];

    Frag16 pf;
    {
      const unsigned* pp = PB + (wv * 16 + ml) * 20 + hi * 4;
      pf.q[0] = *(const uint4*)pp;        // pairs hi*4 + 0..3
      pf.q[1] = *(const uint4*)(pp + 8);  // pairs 8 + hi*4 + 0..3
    }
#pragma unroll
    for (int t = 0; t < 4; ++t) {
      Frag16 vf;
      const unsigned* vpl = VB + (t * 16 + ml) * 20 + hi * 4;
      vf.q[0] = *(const uint4*)vpl;
      vf.q[1] = *(const uint4*)(vpl + 8);
      oacc[t] = wmma_bf16(pf, vf, oacc[t]);
    }
    __syncthreads();
  }

  // normalize and store merged-head bf16 output
  if (lane < 16) LL[wv * 16 + lane] = l_i;
  float lv[8];
#pragma unroll
  for (int r = 0; r < 8; ++r) lv[r] = 1.0f / LL[wv * 16 + r + 8 * hi];
#pragma unroll
  for (int t = 0; t < 4; ++t)
#pragma unroll
    for (int r = 0; r < 8; ++r)
      ob[base + (size_t)(q0 + wv * 16 + r + 8 * hi) * DMODEL + t * 16 + ml] =
          (unsigned short)f2bf(oacc[t][r] * lv[r]);
}

// ---------------------------------------------------------------------------
extern "C" void kernel_launch(void* const* d_in, const int* in_sizes, int n_in,
                              void* d_out, int out_size, void* d_ws, size_t ws_size,
                              hipStream_t stream) {
  const float* x  = (const float*)d_in[0];
  const float* wq = (const float*)d_in[1];
  const float* wk = (const float*)d_in[2];
  const float* wv = (const float*)d_in[3];
  const float* wo = (const float*)d_in[4];

  const int MT = 2 * SEQ;                          // 4096 token rows
  const size_t BUF = (size_t)MT * DMODEL / 2;      // 2M u32 per bf16 buffer
  const size_t WBUF = (size_t)DMODEL / 2 * DMODEL; // 0.5M u32 per packed weight

  unsigned* W   = (unsigned*)d_ws;
  unsigned* xb  = W;              // bf16 x          (8 MB)
  unsigned* qb  = W + BUF;        // bf16 q (rope)   (8 MB)
  unsigned* kb  = W + 2 * BUF;    // bf16 k (rope)   (8 MB)
  unsigned* vpp = W + 3 * BUF;    // v pair-packed   (8 MB)
  unsigned* obu = W + 4 * BUF;    // bf16 attn out   (8 MB)
  unsigned* wqp = W + 5 * BUF;
  unsigned* wkp = wqp + WBUF;
  unsigned* wvp = wkp + WBUF;
  unsigned* wop = wvp + WBUF;                       // total 48 MB

  // one-time bf16 packing
  pack_rows_kernel<<<(int)(BUF / 256), 256, 0, stream>>>(x, xb, (int)BUF);
  pack_weight_kernel<<<(int)(WBUF / 256), 256, 0, stream>>>(wq, wqp, (int)WBUF);
  pack_weight_kernel<<<(int)(WBUF / 256), 256, 0, stream>>>(wk, wkp, (int)WBUF);
  pack_weight_kernel<<<(int)(WBUF / 256), 256, 0, stream>>>(wv, wvp, (int)WBUF);
  pack_weight_kernel<<<(int)(WBUF / 256), 256, 0, stream>>>(wo, wop, (int)WBUF);

  dim3 gg(DMODEL / 64, MT / 64);                    // (16, 64)
  gemm_wmma<<<gg, 128, 0, stream>>>(xb, wqp, qb, 1, MT, DMODEL, DMODEL);
  gemm_wmma<<<gg, 128, 0, stream>>>(xb, wkp, kb, 1, MT, DMODEL, DMODEL);
  gemm_wmma<<<gg, 128, 0, stream>>>(xb, wvp, vpp, 0, MT, DMODEL, DMODEL);
  attn_wmma<<<dim3(SEQ / 64, NHEAD * 2), 128, 0, stream>>>(
      qb, kb, vpp, (unsigned short*)obu);
  gemm_wmma<<<gg, 128, 0, stream>>>(obu, wop, d_out, 2, MT, DMODEL, DMODEL);
}
